// GAU_53987738911203
// MI455X (gfx1250) — matrix operand order repaired and checked by reading, not compiled, not run on hardware
//
#include <hip/hip_runtime.h>

// Problem constants (from the reference)
#define HW    3600            // 60*60
#define CIN   1536
#define COUT  256
#define NB    4
#define PT48  75              // HW / 48  (exact)
#define FINAL_ELEMS (NB * COUT * HW)
#define PADK  40              // LDS row pitch (bf16) for 32 K-entries: 80B keeps 16B
                              // alignment for ds_load_b128 and spreads banks (stride 20 dw)

typedef __attribute__((ext_vector_type(16))) __bf16 bf16x16;  // 8 VGPRs: A/B frag
typedef __attribute__((ext_vector_type(8)))  __bf16 bf16x8;   // 16-byte vector load
typedef __attribute__((ext_vector_type(4)))  __bf16 bf16x4;   // 8-byte vector store
typedef __attribute__((ext_vector_type(8)))  float  f32x8;    // 8 VGPRs: C/D frag
typedef __attribute__((ext_vector_type(4)))  float  f32x4;    // 16-byte vector load/store

__device__ __forceinline__ int lane_id() { return (int)(threadIdx.x & 31u); }
__device__ __forceinline__ int wave_id() { return (int)(threadIdx.x >> 5u); }

__device__ __forceinline__ f32x8 wmma_bf16(bf16x16 a, bf16x16 b, f32x8 c) {
  // (neg_a, A, neg_b, B, c_mod, C, reuse_a, reuse_b)
  return __builtin_amdgcn_wmma_f32_16x16x32_bf16(false, a, false, b, (short)0, c,
                                                 false, false);
}

__device__ __forceinline__ bf16x16 cat8(bf16x8 lo, bf16x8 hi) {
  return __builtin_shufflevector(lo, hi, 0, 1, 2, 3, 4, 5, 6, 7,
                                 8, 9, 10, 11, 12, 13, 14, 15);
}

// ---- fragment layouts (wave32) ---------------------------------------------
// 16-bit A 16x32 (MxK):  lane l: M=l&15, halfK=(l>>4)*8
//   elem e<8 : K = halfK + e        (contiguous run of 8)
//   elem e>=8: K = 16 + halfK + e-8 (contiguous run of 8)
// 16-bit B 32x16 (KxN):  lane l: N=l&15, halfK=(l>>4)*8
//   elem e   : K as for A (e<8: halfK+e, e>=8: 16+halfK+e-8)
// f32  C/D 16x16:        lane l: N=l&15, elem r: M = r + (l>>4)*8

// A[m][k] = src[(m0+m)*ld + k0+K], row-major bf16 — two b128 loads
__device__ __forceinline__ bf16x16 loadA_rm_bf16(const __bf16* __restrict__ src,
                                                 long ld, int m0, int k0) {
  int l = lane_id();
  const __bf16* p = src + (long)(m0 + (l & 15)) * ld + k0 + ((l >> 4) * 8);
  return cat8(*(const bf16x8*)p, *(const bf16x8*)(p + 16));
}

// A[m][k] = src[(m0+m)*ld + k0+K], row-major f32 (unguarded) — four b128 loads
__device__ __forceinline__ bf16x16 loadA_rm_f32(const float* __restrict__ src,
                                                long ld, int m0, int k0) {
  int l = lane_id();
  const float* p = src + (long)(m0 + (l & 15)) * ld + k0 + ((l >> 4) * 8);
  f32x4 v0 = *(const f32x4*)(p);
  f32x4 v1 = *(const f32x4*)(p + 4);
  f32x4 v2 = *(const f32x4*)(p + 16);
  f32x4 v3 = *(const f32x4*)(p + 20);
  bf16x16 a;
#pragma unroll
  for (int e = 0; e < 4; ++e) {
    a[e]      = (__bf16)v0[e];
    a[e + 4]  = (__bf16)v1[e];
    a[e + 8]  = (__bf16)v2[e];
    a[e + 12] = (__bf16)v3[e];
  }
  return a;
}

// A row-major f32, scalar guarded (K tail)
__device__ __forceinline__ bf16x16 loadA_rm_f32_g(const float* __restrict__ src,
                                                  long ld, int m0, int k0, int kmax) {
  int l = lane_id();
  long m = m0 + (l & 15);
  int kh = (l >> 4) * 8;
  bf16x16 a;
#pragma unroll
  for (int e = 0; e < 16; ++e) {
    int K = (e < 8) ? (kh + e) : (16 + kh + (e - 8));
    int kg = k0 + K;
    a[e] = (kg < kmax) ? (__bf16)src[m * ld + kg] : (__bf16)0.0f;
  }
  return a;
}

// B[k][n] = src[(n0+n)*ld + k0+K], bf16 rows indexed by N — two b128 loads
__device__ __forceinline__ bf16x16 loadB_nrow_bf16(const __bf16* __restrict__ src,
                                                   long ld, int k0, int n0) {
  int l = lane_id();
  const __bf16* p = src + (long)(n0 + (l & 15)) * ld + k0 + ((l >> 4) * 8);
  return cat8(*(const bf16x8*)p, *(const bf16x8*)(p + 16));
}

// B nrow bf16, scalar guarded on K (vsT tail)
__device__ __forceinline__ bf16x16 loadB_nrow_bf16_g(const __bf16* __restrict__ src,
                                                     long ld, int k0, int n0, int kmax) {
  int l = lane_id();
  long n = n0 + (l & 15);
  int kh = (l >> 4) * 8;
  bf16x16 b;
#pragma unroll
  for (int e = 0; e < 16; ++e) {
    int kg = k0 + ((e < 8) ? (kh + e) : (16 + kh + (e - 8)));
    b[e] = (kg < kmax) ? src[n * ld + kg] : (__bf16)0.0f;
  }
  return b;
}

// ---- K-1: bulk f32 -> bf16 conversion (weights), 4 elements/thread ----------
__global__ void GAU_cvt_kernel(const float* __restrict__ src, __bf16* __restrict__ dst,
                               int n4) {
  int i = blockIdx.x * blockDim.x + threadIdx.x;
  if (i >= n4) return;
  f32x4 v = ((const f32x4*)src)[i];
  bf16x4 o;
#pragma unroll
  for (int e = 0; e < 4; ++e) o[e] = (__bf16)v[e];
  ((bf16x4*)dst)[i] = o;
}

// ---- K0: resize is exact stride-8 sampling (scale=(473-1)/(60-1)=8) ---------
__global__ void GAU_mask_kernel(const float* __restrict__ mask, float* __restrict__ bias) {
  int idx = blockIdx.x * blockDim.x + threadIdx.x;   // b*HW + p
  if (idx >= NB * HW) return;
  int b = idx / HW, p = idx - b * HW;
  int y = p / 60, x = p - y * 60;
  float fg = mask[((long)b * 473 + y * 8) * 473 + x * 8];
  bias[idx] = (1.0f - fg) * -999.0f;
}

// ---- K1: proj = sum_c X[b][c][p]*W[o][c]; optional Y[b][p][o] and YT[b][o][p]
// 8 waves/block, wave tile 48(p) x 32(o). The 32Kx48P f32 A-tile is staged once
// per block through LDS (coalesced global b32 reads, bf16 [p][K] layout), then
// every wave builds A fragments with two ds_load_b128 each. 6 WMMA per K-step.
__global__ void GAU_proj_kernel(const float* __restrict__ X, const __bf16* __restrict__ Wb,
                                __bf16* __restrict__ Y, __bf16* __restrict__ YT) {
  __shared__ __bf16 sA[48][PADK];                  // 3840 B
  int b  = blockIdx.z;
  int p0 = blockIdx.x * 48;
  int o0 = wave_id() * 32;                         // 8 waves * 32 = 256 outs
  const float* Xb = X + (long)b * CIN * HW;
  int l = lane_id();
  int t = (int)threadIdx.x;
  f32x8 acc[3][2] = {};
  for (int k0 = 0; k0 < CIN; k0 += 32) {
    // cooperative stage: 32(K) x 48(p), 1536 f32, 6 per thread, coalesced in p
#pragma unroll
    for (int rep = 0; rep < 6; ++rep) {
      int idx = rep * 256 + t;                     // 0..1535
      int kr  = idx / 48;                          // K row 0..31
      int pc  = idx - kr * 48;                     // p col 0..47
      sA[pc][kr] = (__bf16)Xb[(long)(k0 + kr) * HW + p0 + pc];
    }
    __syncthreads();
    bf16x16 A[3], Bf[2];
#pragma unroll
    for (int i = 0; i < 3; ++i) {                  // two ds_load_b128 per fragment
      const __bf16* pa = &sA[16 * i + (l & 15)][(l >> 4) * 8];
      A[i] = cat8(*(const bf16x8*)pa, *(const bf16x8*)(pa + 16));
    }
#pragma unroll
    for (int j = 0; j < 2; ++j) Bf[j] = loadB_nrow_bf16(Wb, CIN, k0, o0 + 16 * j);
#pragma unroll
    for (int i = 0; i < 3; ++i)
#pragma unroll
      for (int j = 0; j < 2; ++j) acc[i][j] = wmma_bf16(A[i], Bf[j], acc[i][j]);
    __syncthreads();                               // protect sA before next stage
  }
  long bp = (long)b * HW;
#pragma unroll
  for (int i = 0; i < 3; ++i) {
    int rbase = p0 + 16 * i + ((l >> 4) << 3);
#pragma unroll
    for (int j = 0; j < 2; ++j) {
      int col = o0 + 16 * j + (l & 15);
#pragma unroll
      for (int r = 0; r < 8; ++r) {
        __bf16 v = (__bf16)acc[i][j][r];
        if (Y)  Y[(bp + rbase + r) * COUT + col] = v;
        if (YT) YT[((long)b * COUT + col) * HW + rbase + r] = v;
      }
    }
  }
}

// ---- K2: attn[b][p][j] = q[p]·k[j] + bias[b][j]  (f32, pre-softmax) ---------
// 4 waves/block, wave tile 48 x 48 (all-bf16): 3 A + 3 B frags, 9 WMMA per K-step
__global__ void GAU_scores_kernel(const __bf16* __restrict__ q, const __bf16* __restrict__ k,
                                  const float* __restrict__ bias, float* __restrict__ attn) {
  int b  = blockIdx.z;
  int p0 = blockIdx.x * 48;
  int jt = blockIdx.y * 4 + wave_id();   // wave-uniform guard: EXEC stays all-1s
  if (jt >= PT48) return;
  int j0 = jt * 48;
  const __bf16* qb = q + (long)b * HW * COUT;
  const __bf16* kb = k + (long)b * HW * COUT;
  int l = lane_id();
  f32x8 acc[3][3] = {};
#pragma unroll
  for (int k0 = 0; k0 < COUT; k0 += 32) {
    bf16x16 A[3], Bf[3];
#pragma unroll
    for (int i = 0; i < 3; ++i) A[i] = loadA_rm_bf16(qb, COUT, p0 + 16 * i, k0);
#pragma unroll
    for (int j = 0; j < 3; ++j) Bf[j] = loadB_nrow_bf16(kb, COUT, k0, j0 + 16 * j);
#pragma unroll
    for (int i = 0; i < 3; ++i)
#pragma unroll
      for (int j = 0; j < 3; ++j) acc[i][j] = wmma_bf16(A[i], Bf[j], acc[i][j]);
  }
  float* arow = attn + (long)b * HW * HW;
#pragma unroll
  for (int j = 0; j < 3; ++j) {
    int col = j0 + 16 * j + (l & 15);
    float bv = bias[b * HW + col];
#pragma unroll
    for (int i = 0; i < 3; ++i) {
      int rbase = p0 + 16 * i + ((l >> 4) << 3);
#pragma unroll
      for (int r = 0; r < 8; ++r)
        arow[(long)(rbase + r) * HW + col] = acc[i][j][r] + bv;
    }
  }
}

// ---- K3: in-place row softmax over 3600 (900 x float4) ----------------------
__global__ void GAU_softmax_kernel(float* __restrict__ attn) {
  long row = blockIdx.x;                 // b*HW + p
  f32x4* p4 = (f32x4*)(attn + row * HW); // 900 vectors
  __shared__ float red[256];
  int t = threadIdx.x;
  float m = -3.0e38f;
  for (int i = t; i < 900; i += 256) {
    f32x4 v = p4[i];
    m = fmaxf(m, fmaxf(fmaxf(v[0], v[1]), fmaxf(v[2], v[3])));
  }
  red[t] = m; __syncthreads();
  for (int s = 128; s > 0; s >>= 1) {
    if (t < s) red[t] = fmaxf(red[t], red[t + s]);
    __syncthreads();
  }
  m = red[0]; __syncthreads();
  float sum = 0.0f;
  for (int i = t; i < 900; i += 256) {
    f32x4 v = p4[i];
#pragma unroll
    for (int e = 0; e < 4; ++e) { v[e] = __expf(v[e] - m); sum += v[e]; }
    p4[i] = v;
  }
  red[t] = sum; __syncthreads();
  for (int s = 128; s > 0; s >>= 1) {
    if (t < s) red[t] += red[t + s];
    __syncthreads();
  }
  float inv = 1.0f / red[0];
  for (int i = t; i < 900; i += 256) { f32x4 v = p4[i]; v *= inv; p4[i] = v; }
}

// ---- K4: out[b][p][d] = sum_j attn[p][j] * vsT[d][j]  (bf16 out) ------------
// 8 waves/block, wave tile 48(p) x 32(d): 3 A(f32) + 2 B frags, 6 WMMA per K-step
__global__ void GAU_av_kernel(const float* __restrict__ attn, const __bf16* __restrict__ vsT,
                              __bf16* __restrict__ outb) {
  int b  = blockIdx.z;
  int p0 = blockIdx.x * 48;
  int d0 = wave_id() * 32;                        // 8 waves * 32 = 256 dims
  const float*  ab = attn + (long)b * HW * HW;
  const __bf16* vb = vsT + (long)b * COUT * HW;   // [d][p] layout, ld = HW
  int l = lane_id();
  f32x8 acc[3][2] = {};
  for (int k0 = 0; k0 + 32 <= HW; k0 += 32) {     // 112 full steps
    __builtin_prefetch(ab + (long)(p0 + (l & 15)) * HW + k0 + 128, 0, 0);
    bf16x16 A[3], Bf[2];
#pragma unroll
    for (int i = 0; i < 3; ++i) A[i] = loadA_rm_f32(ab, HW, p0 + 16 * i, k0);
#pragma unroll
    for (int j = 0; j < 2; ++j) Bf[j] = loadB_nrow_bf16(vb, HW, k0, d0 + 16 * j);
#pragma unroll
    for (int i = 0; i < 3; ++i)
#pragma unroll
      for (int j = 0; j < 2; ++j) acc[i][j] = wmma_bf16(A[i], Bf[j], acc[i][j]);
  }
  {                                               // tail: k0 = 3584, 16 valid K
    const int k0 = (HW / 32) * 32;
    bf16x16 A[3], Bf[2];
#pragma unroll
    for (int i = 0; i < 3; ++i) A[i] = loadA_rm_f32_g(ab, HW, p0 + 16 * i, k0, HW);
#pragma unroll
    for (int j = 0; j < 2; ++j) Bf[j] = loadB_nrow_bf16_g(vb, HW, k0, d0 + 16 * j, HW);
#pragma unroll
    for (int i = 0; i < 3; ++i)
#pragma unroll
      for (int j = 0; j < 2; ++j) acc[i][j] = wmma_bf16(A[i], Bf[j], acc[i][j]);
  }
  long bp = (long)b * HW;
#pragma unroll
  for (int i = 0; i < 3; ++i) {
    int rbase = p0 + 16 * i + ((l >> 4) << 3);
#pragma unroll
    for (int j = 0; j < 2; ++j) {
      int col = d0 + 16 * j + (l & 15);
#pragma unroll
      for (int r = 0; r < 8; ++r)
        outb[(bp + rbase + r) * COUT + col] = (__bf16)acc[i][j][r];
    }
  }
}

// ---- K5: final[b][o][p] = relu( Wf[o][:]·concat(out[p], vq[p]) + bias[o] ) --
// 8 waves/block, wave tile 48(p) x 32(o), K=512 (windows never straddle concat)
__global__ void GAU_final_kernel(const __bf16* __restrict__ outb, const __bf16* __restrict__ vq,
                                 const __bf16* __restrict__ Wfb, const float* __restrict__ bfin,
                                 float* __restrict__ final_out) {
  int b  = blockIdx.z;
  int p0 = blockIdx.x * 48;
  int o0 = wave_id() * 32;
  const __bf16* ob = outb + (long)b * HW * COUT;
  const __bf16* vb = vq + (long)b * HW * COUT;
  int l = lane_id();
  f32x8 acc[3][2] = {};
#pragma unroll
  for (int k0 = 0; k0 < 2 * COUT; k0 += 32) {
    const __bf16* srcA = (k0 < COUT) ? ob : vb;   // window-uniform concat source
    int koff = k0 & (COUT - 1);
    bf16x16 A[3], Bf[2];
#pragma unroll
    for (int i = 0; i < 3; ++i) A[i] = loadA_rm_bf16(srcA, COUT, p0 + 16 * i, koff);
#pragma unroll
    for (int j = 0; j < 2; ++j) Bf[j] = loadB_nrow_bf16(Wfb, 2 * COUT, k0, o0 + 16 * j);
#pragma unroll
    for (int i = 0; i < 3; ++i)
#pragma unroll
      for (int j = 0; j < 2; ++j) acc[i][j] = wmma_bf16(A[i], Bf[j], acc[i][j]);
  }
#pragma unroll
  for (int j = 0; j < 2; ++j) {
    int o = o0 + 16 * j + (l & 15);
    float bv = bfin[o];
    float* orow = final_out + (long)(b * COUT + o) * HW;
#pragma unroll
    for (int i = 0; i < 3; ++i) {
      int rbase = p0 + 16 * i + ((l >> 4) << 3);
      f32x4 lo, hi;                                  // rows are contiguous in [o][p]
#pragma unroll
      for (int r = 0; r < 4; ++r) {
        float v0 = acc[i][j][r] + bv;
        float v1 = acc[i][j][r + 4] + bv;
        lo[r] = v0 > 0.0f ? v0 : 0.0f;
        hi[r] = v1 > 0.0f ? v1 : 0.0f;
      }
      *(f32x4*)(orow + rbase)     = lo;
      *(f32x4*)(orow + rbase + 4) = hi;
    }
  }
}

extern "C" void kernel_launch(void* const* d_in, const int* in_sizes, int n_in,
                              void* d_out, int out_size, void* d_ws, size_t ws_size,
                              hipStream_t stream) {
  const float* feat_supp  = (const float*)d_in[0];
  const float* feat_query = (const float*)d_in[1];
  const float* mask_supp  = (const float*)d_in[2];
  const float* W_lpf      = (const float*)d_in[3];
  const float* W_supp     = (const float*)d_in[4];
  const float* W_query    = (const float*)d_in[5];
  const float* W_final    = (const float*)d_in[6];
  const float* b_final    = (const float*)d_in[7];

  float* final_out = (float*)d_out;
  float* attn      = final_out + FINAL_ELEMS;   // [B][hw][hw] f32 region of d_out

  // workspace (~40 MB): 5 bf16 activation tensors + bf16 weights + f32 bias
  const size_t S  = (size_t)NB * HW * COUT;     // activation tensor elems
  const size_t SW = (size_t)COUT * CIN;         // 1x1 weight elems
  __bf16* q_bf   = (__bf16*)d_ws;
  __bf16* k_bf   = q_bf   + S;
  __bf16* vsT_bf = k_bf   + S;                  // [B][256][hw] (transposed V)
  __bf16* vq_bf  = vsT_bf + S;
  __bf16* out_bf = vq_bf  + S;
  __bf16* Wlpf_b = out_bf + S;
  __bf16* Wsup_b = Wlpf_b + SW;
  __bf16* Wqry_b = Wsup_b + SW;
  __bf16* Wfin_b = Wqry_b + SW;                 // [256][512]
  float*  bias   = (float*)(Wfin_b + (size_t)COUT * 2 * COUT);

  // ---- weight conversion + mask bias (tiny) ----
  const int n4w = (COUT * CIN) / 4, n4f = (COUT * 2 * COUT) / 4;
  GAU_cvt_kernel<<<dim3((n4w + 255) / 256), dim3(256), 0, stream>>>(W_lpf,   Wlpf_b, n4w);
  GAU_cvt_kernel<<<dim3((n4w + 255) / 256), dim3(256), 0, stream>>>(W_supp,  Wsup_b, n4w);
  GAU_cvt_kernel<<<dim3((n4w + 255) / 256), dim3(256), 0, stream>>>(W_query, Wqry_b, n4w);
  GAU_cvt_kernel<<<dim3((n4f + 255) / 256), dim3(256), 0, stream>>>(W_final, Wfin_b, n4f);
  GAU_mask_kernel<<<dim3((NB * HW + 255) / 256), dim3(256), 0, stream>>>(mask_supp, bias);

  dim3 blk8(256);                               // 8 waves (wave32)
  dim3 blk4(128);                               // 4 waves

  dim3 gproj(PT48, 1, NB);                      // 75 p-tiles, 8 waves -> 256 o
  GAU_proj_kernel<<<gproj, blk8, 0, stream>>>(feat_supp,  Wsup_b, k_bf,  (__bf16*)nullptr);
  GAU_proj_kernel<<<gproj, blk8, 0, stream>>>(feat_query, Wqry_b, q_bf,  (__bf16*)nullptr);
  GAU_proj_kernel<<<gproj, blk8, 0, stream>>>(feat_supp,  Wlpf_b, (__bf16*)nullptr, vsT_bf);
  GAU_proj_kernel<<<gproj, blk8, 0, stream>>>(feat_query, Wlpf_b, vq_bf, (__bf16*)nullptr);

  GAU_scores_kernel<<<dim3(PT48, (PT48 + 3) / 4, NB), blk4, 0, stream>>>(q_bf, k_bf, bias,
                                                                         attn);

  GAU_softmax_kernel<<<dim3(NB * HW), dim3(256), 0, stream>>>(attn);

  GAU_av_kernel<<<dim3(PT48, 1, NB), blk8, 0, stream>>>(attn, vsT_bf, out_bf);

  GAU_final_kernel<<<dim3(PT48, 1, NB), blk8, 0, stream>>>(out_bf, vq_bf, Wfin_b, b_final,
                                                           final_out);
}
